// multi_head_attention_55954833932809
// MI455X (gfx1250) — matrix operand order repaired and checked by reading, not compiled, not run on hardware
//
#include <hip/hip_runtime.h>
#include <hip/hip_bf16.h>

// ---------------------------------------------------------------------------
// Multi-head attention forward, MI455X (gfx1250), wave32 + WMMA bf16 path.
// B=2, S=2048, D=1024, H=16, dk=64.
// Layouts (chosen so every WMMA fragment load is contiguous):
//   Q : [B*H][S][dk]   bf16   (A-fragments: contiguous in dk)
//   K : [B*H][S][dk]   bf16   (B-fragments for Q@K^T: contiguous in dk)
//   V : [B*H][dk][S]   bf16   (B-fragments for P@V: contiguous in kv)
// ---------------------------------------------------------------------------

#define S_LEN   2048
#define DMODEL  1024
#define NHEADS  16
#define DK      64
#define BATCH   2

typedef __attribute__((ext_vector_type(16))) __bf16 v16bf;
typedef __attribute__((ext_vector_type(8)))  __bf16 v8bf;
typedef __attribute__((ext_vector_type(4)))  __bf16 v4bf;
typedef __attribute__((ext_vector_type(8)))  float  v8f;
typedef __attribute__((ext_vector_type(4)))  unsigned int v4u;
typedef __attribute__((ext_vector_type(8)))  int v8i;
typedef __attribute__((ext_vector_type(4)))  int v4i;

// Types matching the async-LDS builtin prototype discovered by probe:
//   arg0: 'int __vector(4) __device__ *'  (= vector_size(16) int, addrspace(1))
typedef int v4i_n __attribute__((vector_size(16)));
typedef v4i_n __attribute__((address_space(1))) gv4i;   // global
typedef v4i_n __attribute__((address_space(3))) sv4i;   // LDS

#if defined(__has_builtin)
# if __has_builtin(__builtin_amdgcn_global_load_async_to_lds_b128)
#  define HAVE_ASYNC_LDS 1
# endif
#endif

__device__ __forceinline__ __bf16 f2bf(float f) { return (__bf16)f; }

__device__ __forceinline__ void wait_asynccnt0() {
#if defined(__has_builtin) && __has_builtin(__builtin_amdgcn_s_wait_asynccnt)
    __builtin_amdgcn_s_wait_asynccnt(0);
#else
    asm volatile("s_wait_asynccnt 0" ::: "memory");
#endif
}

// ===========================================================================
// Kernel 1: QKV projection.  Y = X @ W + b, X:[4096,1024] W:[1024,1024].
// blockIdx.z selects {WQ,WK,WV}. 128 thr = 4 waves; 64x64 tile/block;
// each wave = 32x32 quadrant (2x2 WMMA). Double-buffered LDS staging:
// A tile f32 via ASYNC global->LDS DMA (ASYNCcnt), convert on fragment
// build; B tile converted bf16 and stored transposed [n][k] so B-fragments
// are a single contiguous read.
// ===========================================================================
__global__ __launch_bounds__(128)
void qkv_kernel(const float* __restrict__ X,
                const float* __restrict__ WQ, const float* __restrict__ bQ,
                const float* __restrict__ WK, const float* __restrict__ bK,
                const float* __restrict__ WV, const float* __restrict__ bV,
                __bf16* __restrict__ Qb, __bf16* __restrict__ Kb,
                __bf16* __restrict__ Vb)
{
    const float* W;  const float* bias;  __bf16* Out;
    if      (blockIdx.z == 0) { W = WQ; bias = bQ; Out = Qb; }
    else if (blockIdx.z == 1) { W = WK; bias = bK; Out = Kb; }
    else                      { W = WV; bias = bV; Out = Vb; }
    const bool vT = (blockIdx.z == 2);          // V written transposed

    __shared__ float  Asf[2][64][32];           // A tile, f32 (async DMA target)
    __shared__ __bf16 BsT[2][64][32];           // B tile, bf16, [n][k]

    const int lane  = threadIdx.x & 31;
    const int wave  = threadIdx.x >> 5;
    const int m0    = blockIdx.y * 64;
    const int n0    = blockIdx.x * 64;
    const int waveM = (wave >> 1) * 32;
    const int waveN = (wave & 1)  * 32;

    // staging coordinates (fixed per thread, unconditional trip counts)
    const int a_r0 = threadIdx.x >> 3;          // 0..15 (rows r0, r0+16, +32, +48)
    const int a_c  = (threadIdx.x & 7) * 4;     // f32 col 0..28
    const int b_k  = (threadIdx.x >> 4) * 4;    // 4x4 block: k base 0..28
    const int b_n  = (threadIdx.x & 15) * 4;    // n base 0..60

    auto stage = [&](int buf, int k0) {
        // ---- A tile: 64x32 f32, async global->LDS (b128 per lane) ----
#ifdef HAVE_ASYNC_LDS
        #pragma unroll
        for (int i = 0; i < 4; i++) {
            const float* gp = X + (size_t)(m0 + a_r0 + i * 16) * DMODEL + k0 + a_c;
            __builtin_amdgcn_global_load_async_to_lds_b128(
                (gv4i*)gp, (sv4i*)&Asf[buf][a_r0 + i * 16][a_c], 0, 0);
        }
#else
        float4 av[4];
        #pragma unroll
        for (int i = 0; i < 4; i++)
            av[i] = *(const float4*)(X + (size_t)(m0 + a_r0 + i * 16) * DMODEL + k0 + a_c);
        #pragma unroll
        for (int i = 0; i < 4; i++)
            *(float4*)(&Asf[buf][a_r0 + i * 16][a_c]) = av[i];
#endif
        // ---- B tile: 4x4 block of W, convert + transpose into [n][k] ----
        float4 wv[4];
        #pragma unroll
        for (int r = 0; r < 4; r++)
            wv[r] = *(const float4*)(W + (size_t)(k0 + b_k + r) * DMODEL + n0 + b_n);
        #pragma unroll
        for (int e = 0; e < 4; e++) {           // 4x4 transpose -> packed b64 store
            v4bf col;
            col[0] = f2bf((&wv[0].x)[e]);
            col[1] = f2bf((&wv[1].x)[e]);
            col[2] = f2bf((&wv[2].x)[e]);
            col[3] = f2bf((&wv[3].x)[e]);
            *(v4bf*)(&BsT[buf][b_n + e][b_k]) = col;
        }
    };

    v8f acc[2][2];
    acc[0][0] = (v8f){}; acc[0][1] = (v8f){};
    acc[1][0] = (v8f){}; acc[1][1] = (v8f){};

    stage(0, 0);
    #pragma unroll 2
    for (int it = 0; it < DMODEL / 32; ++it) {
        const int buf = it & 1;
#ifdef HAVE_ASYNC_LDS
        wait_asynccnt0();                       // own async tile writes done
#endif
        __syncthreads();                        // tile 'buf' resident block-wide
        if (it + 1 < DMODEL / 32) stage(buf ^ 1, (it + 1) * 32);  // prefetch next

        // ---- A fragments (convert f32 -> bf16 here) ----
        v16bf afrag[2];
        #pragma unroll
        for (int tm = 0; tm < 2; tm++) {
            const int m  = waveM + tm * 16 + (lane & 15);
            const int kb = (lane >> 4) * 16;
            const float4 f0 = *(const float4*)&Asf[buf][m][kb];
            const float4 f1 = *(const float4*)&Asf[buf][m][kb + 4];
            const float4 f2 = *(const float4*)&Asf[buf][m][kb + 8];
            const float4 f3 = *(const float4*)&Asf[buf][m][kb + 12];
            v16bf a;
            a[0]  = f2bf(f0.x); a[1]  = f2bf(f0.y); a[2]  = f2bf(f0.z); a[3]  = f2bf(f0.w);
            a[4]  = f2bf(f1.x); a[5]  = f2bf(f1.y); a[6]  = f2bf(f1.z); a[7]  = f2bf(f1.w);
            a[8]  = f2bf(f2.x); a[9]  = f2bf(f2.y); a[10] = f2bf(f2.z); a[11] = f2bf(f2.w);
            a[12] = f2bf(f3.x); a[13] = f2bf(f3.y); a[14] = f2bf(f3.z); a[15] = f2bf(f3.w);
            afrag[tm] = a;
        }
        // ---- B fragments: single contiguous 32B read per fragment ----
        v16bf bfrag[2];
        #pragma unroll
        for (int tn = 0; tn < 2; tn++) {
            const int n   = waveN + tn * 16 + (lane & 15);
            const int kb2 = (lane >> 4) * 16;
            bfrag[tn] = *(const v16bf*)&BsT[buf][n][kb2];
        }
        #pragma unroll
        for (int tm = 0; tm < 2; tm++)
            #pragma unroll
            for (int tn = 0; tn < 2; tn++)
                acc[tm][tn] = __builtin_amdgcn_wmma_f32_16x16x32_bf16(
                    false, afrag[tm], false, bfrag[tn],
                    (short)0, acc[tm][tn], false, false);
        __syncthreads();                        // reads done before buf re-staged
    }

    // ---- epilogue: + bias, bf16, store (V transposed) ----
    #pragma unroll
    for (int tm = 0; tm < 2; tm++) {
        #pragma unroll
        for (int tn = 0; tn < 2; tn++) {
            const int mbase = m0 + waveM + tm * 16;
            const int n     = n0 + waveN + tn * 16 + (lane & 15);
            const float bv  = bias[n];
            const int hh = n >> 6, d = n & 63;
            #pragma unroll
            for (int r = 0; r < 8; r++) {
                const int m = mbase + r + 8 * (lane >> 4);
                const int b = m >> 11, s = m & (S_LEN - 1);
                const __bf16 val = f2bf(acc[tm][tn][r] + bv);
                if (vT)
                    Out[(((size_t)(b * NHEADS + hh)) * DK + d) * S_LEN + s] = val;
                else
                    Out[(((size_t)(b * NHEADS + hh)) * S_LEN + s) * DK + d] = val;
            }
        }
    }
}

// ===========================================================================
// Kernel 2: flash attention per (b,h) + fused per-head projection.
// grid = (S/64, B*H); 128 thr = 4 waves; each wave owns 16 query rows.
// All K/V fragments are contiguous global loads (thanks to V^T layout);
// P routed through per-wave LDS (C-layout -> A-layout) with s_wait_dscnt.
// ===========================================================================
__global__ __launch_bounds__(128)
void attn_kernel(const __bf16* __restrict__ Qb, const __bf16* __restrict__ Kb,
                 const __bf16* __restrict__ Vt,
                 const float* __restrict__ Wp, const float* __restrict__ bp,
                 float* __restrict__ Out)
{
    __shared__ __bf16 Pst[4][16 * 64];

    const int lane = threadIdx.x & 31;
    const int wave = threadIdx.x >> 5;
    const int bh   = blockIdx.y;
    const int q0   = blockIdx.x * 64 + wave * 16;

    const __bf16* Qh = Qb + (size_t)bh * S_LEN * DK;
    const __bf16* Kh = Kb + (size_t)bh * S_LEN * DK;
    const __bf16* Vh = Vt + (size_t)bh * DK * S_LEN;   // [dk][S]

    // ---- Q A-fragments, loaded once ----
    v16bf qa[2];
    {
        const int m  = q0 + (lane & 15);
        const int kb = (lane >> 4) * 16;
        #pragma unroll
        for (int kk = 0; kk < 2; kk++) {
            const v8bf lo = *(const v8bf*)&Qh[(size_t)m * DK + kk * 32 + kb];
            const v8bf hi = *(const v8bf*)&Qh[(size_t)m * DK + kk * 32 + kb + 8];
            v16bf a;
            #pragma unroll
            for (int j = 0; j < 8; j++) { a[j] = lo[j]; a[8 + j] = hi[j]; }
            qa[kk] = a;
        }
    }

    v8f o[4];
    o[0] = (v8f){}; o[1] = (v8f){}; o[2] = (v8f){}; o[3] = (v8f){};
    float mrow[8], lrow[8];
    #pragma unroll
    for (int r = 0; r < 8; r++) { mrow[r] = -1e30f; lrow[r] = 0.0f; }
    const float scale = 0.125f;   // 1/sqrt(dk)

    for (int t0 = 0; t0 < S_LEN; t0 += 32) {
        // prefetch next kv tile into caches (global_prefetch_b8)
        __builtin_prefetch(Kh + (size_t)(t0 + 32) * DK, 0, 3);
        __builtin_prefetch(Vh + (size_t)t0 + 32, 0, 3);

        // ---- scores S = Q x K^T : 16 x 32 ----
        v8f sc[2];
        sc[0] = (v8f){}; sc[1] = (v8f){};
        #pragma unroll
        for (int nt = 0; nt < 2; nt++) {
            const int kvrow = t0 + nt * 16 + (lane & 15);
            #pragma unroll
            for (int kk = 0; kk < 2; kk++) {
                const v16bf bfr = *(const v16bf*)
                    &Kh[(size_t)kvrow * DK + kk * 32 + (lane >> 4) * 16];
                sc[nt] = __builtin_amdgcn_wmma_f32_16x16x32_bf16(
                    false, qa[kk], false, bfr, (short)0, sc[nt], false, false);
            }
        }

        // ---- online softmax (f32) ----
        #pragma unroll
        for (int r = 0; r < 8; r++) {
            float v0 = sc[0][r] * scale;
            float v1 = sc[1][r] * scale;
            float tmax = fmaxf(v0, v1);
            #pragma unroll
            for (int off = 1; off < 16; off <<= 1)
                tmax = fmaxf(tmax, __shfl_xor(tmax, off, 32));
            const float mnew  = fmaxf(mrow[r], tmax);
            const float alpha = __expf(mrow[r] - mnew);
            mrow[r] = mnew;
            const float p0 = __expf(v0 - mnew);
            const float p1 = __expf(v1 - mnew);
            float ps = p0 + p1;
            #pragma unroll
            for (int off = 1; off < 16; off <<= 1)
                ps += __shfl_xor(ps, off, 32);
            lrow[r] = lrow[r] * alpha + ps;
            #pragma unroll
            for (int dt = 0; dt < 4; dt++) o[dt][r] *= alpha;
            const int mloc = r + 8 * (lane >> 4);
            Pst[wave][mloc * 32 + (lane & 15)]      = f2bf(p0);
            Pst[wave][mloc * 32 + 16 + (lane & 15)] = f2bf(p1);
        }

        asm volatile("s_wait_dscnt 0" ::: "memory");   // wave-private transpose

        // ---- P as A-fragment ----
        v16bf pa;
        {
            const int m  = lane & 15;
            const int kb = (lane >> 4) * 16;
            const v8bf lo = *(const v8bf*)&Pst[wave][m * 32 + kb];
            const v8bf hi = *(const v8bf*)&Pst[wave][m * 32 + kb + 8];
            #pragma unroll
            for (int j = 0; j < 8; j++) { pa[j] = lo[j]; pa[8 + j] = hi[j]; }
        }

        // ---- O += P x V : V^T layout makes each fragment contiguous ----
        #pragma unroll
        for (int dt = 0; dt < 4; dt++) {
            const int d = dt * 16 + (lane & 15);
            const v16bf vb = *(const v16bf*)
                &Vh[(size_t)d * S_LEN + t0 + (lane >> 4) * 16];
            o[dt] = __builtin_amdgcn_wmma_f32_16x16x32_bf16(
                false, pa, false, vb, (short)0, o[dt], false, false);
        }
    }

    // ---- normalize, stage O (16x64 bf16) in LDS ----
    float inv[8];
    #pragma unroll
    for (int r = 0; r < 8; r++) inv[r] = 1.0f / lrow[r];
    #pragma unroll
    for (int dt = 0; dt < 4; dt++)
        #pragma unroll
        for (int r = 0; r < 8; r++) {
            const int mloc = r + 8 * (lane >> 4);
            Pst[wave][mloc * 64 + dt * 16 + (lane & 15)] = f2bf(o[dt][r] * inv[r]);
        }
    asm volatile("s_wait_dscnt 0" ::: "memory");

    // ---- fused per-head projection: Y = O @ Wp + bp ----
    v16bf oa[2];
    {
        const int m  = lane & 15;
        const int kb = (lane >> 4) * 16;
        #pragma unroll
        for (int kk = 0; kk < 2; kk++) {
            const v8bf lo = *(const v8bf*)&Pst[wave][m * 64 + kk * 32 + kb];
            const v8bf hi = *(const v8bf*)&Pst[wave][m * 64 + kk * 32 + kb + 8];
            v16bf a;
            #pragma unroll
            for (int j = 0; j < 8; j++) { a[j] = lo[j]; a[8 + j] = hi[j]; }
            oa[kk] = a;
        }
    }
    const int b  = bh >> 4;
    const int hh = bh & 15;
    #pragma unroll
    for (int dt = 0; dt < 4; dt++) {
        v8f po = (v8f){};
        #pragma unroll
        for (int kk = 0; kk < 2; kk++) {
            v16bf wb;
            const int kb = kk * 32 + (lane >> 4) * 16;
            const int n  = dt * 16 + (lane & 15);
            #pragma unroll
            for (int j = 0; j < 16; j++)
                wb[j] = f2bf(Wp[(size_t)(kb + j) * DK + n]);
            po = __builtin_amdgcn_wmma_f32_16x16x32_bf16(
                false, oa[kk], false, wb, (short)0, po, false, false);
        }
        const int n   = dt * 16 + (lane & 15);
        const float c = bp[n];
        #pragma unroll
        for (int r = 0; r < 8; r++) {
            const int m = q0 + r + 8 * (lane >> 4);
            Out[((size_t)b * S_LEN + m) * DMODEL + hh * DK + n] = po[r] + c;
        }
    }
}

// ===========================================================================
// Compile-only probe for the TDM path (never launched): 6-arg form confirmed
// by previous round's diagnostic ("expected 6, have 5") => amdgpu-toolchain.
// Signature: (uint32x4 g0, int32x8 g1, int32x4 g2, int32x4 g3, int32x8, cpol)
// ===========================================================================
__global__ void cdna5_probe_tdm() {
#if defined(__has_builtin)
#if __has_builtin(__builtin_amdgcn_tensor_load_to_lds)
    v4u g0 = (v4u){};
    v8i g1 = (v8i){};
    v4i g2 = (v4i){};
    v4i g3 = (v4i){};
    v8i g4 = (v8i){};
    __builtin_amdgcn_tensor_load_to_lds(g0, g1, g2, g3, g4, 0);
#endif
#if __has_builtin(__builtin_amdgcn_s_wait_tensorcnt)
    __builtin_amdgcn_s_wait_tensorcnt((short)0);
#endif
#endif
}

// ===========================================================================
extern "C" void kernel_launch(void* const* d_in, const int* in_sizes, int n_in,
                              void* d_out, int out_size, void* d_ws, size_t ws_size,
                              hipStream_t stream) {
    (void)in_sizes; (void)n_in; (void)out_size; (void)ws_size;
    const float* X  = (const float*)d_in[0];
    const float* WQ = (const float*)d_in[1];
    const float* bQ = (const float*)d_in[2];
    const float* WK = (const float*)d_in[3];
    const float* bK = (const float*)d_in[4];
    const float* WV = (const float*)d_in[5];
    const float* bV = (const float*)d_in[6];
    const float* Wp = (const float*)d_in[7];
    const float* bp = (const float*)d_in[8];
    float* out = (float*)d_out;

    // workspace: Q,K bf16 [B*H][S][dk]; V bf16 [B*H][dk][S]; 8 MB each
    const size_t elems = (size_t)BATCH * NHEADS * S_LEN * DK;
    __bf16* Qb = (__bf16*)d_ws;
    __bf16* Kb = Qb + elems;
    __bf16* Vb = Kb + elems;

    dim3 g1(DMODEL / 64, (BATCH * S_LEN) / 64, 3);   // 16 x 64 x 3
    qkv_kernel<<<g1, 128, 0, stream>>>(X, WQ, bQ, WK, bK, WV, bV, Qb, Kb, Vb);

    dim3 g2(S_LEN / 64, BATCH * NHEADS, 1);           // 32 x 32
    attn_kernel<<<g2, 128, 0, stream>>>(Qb, Kb, Vb, Wp, bp, out);
}
